// SpikingConv2D_47193100649046
// MI455X (gfx1250) — matrix operand order repaired and checked by reading, not compiled, not run on hardware
//
#include <hip/hip_runtime.h>

// SpikingConv2D on MI455X (gfx1250).
// GEMM per 16x16 tile: M=16 x-pixels, N=16 out-channels, K=49 (pad 64).
// f16 split precision: x=hi+lo (11+11 mantissa bits); x*w ~= hi*whi + hi*wlo + lo*whi
// (3 WMMAs/K-step, ~2^-24 relative error -> fp32-grade accuracy).
// Output 805MB fp32 -> HBM-write-bound (~35us @ 23.3TB/s). Weights pre-split into
// B-fragment-ready f16 hi/lo in d_ws (96KB, L2-resident). Patch pre-split at LDS
// staging; A-fragments are ds_load_b32 (const offsets) + v_perm only. D tile has
// M=pixels so each lane stores 8 consecutive floats (2x global_store_b128); the
// fused bias is pre-loaded into the WMMA accumulator.

typedef _Float16 f16;
typedef __attribute__((ext_vector_type(16))) _Float16     v16h;
typedef __attribute__((ext_vector_type(8)))  float        v8f;
typedef __attribute__((ext_vector_type(8)))  unsigned int v8u;

#define H_   256
#define W_   256
#define F_   384
#define KSZ  49
#define KW_  7
#define WS_B_V8U (24 * 2 * 32)   // per array: ft(24) x kk(2) x lane(32) entries of 32B
                                 // hi array then lo array: 2 * 49152 B = 96 KB total

static __device__ __forceinline__ unsigned pack_hilo(float x) {
    // low 16 bits = f16(x), high 16 bits = f16(x - hi)
    const f16 h = (f16)x;
    const f16 l = (f16)(x - (float)h);
    return (unsigned)__builtin_bit_cast(unsigned short, h) |
           ((unsigned)__builtin_bit_cast(unsigned short, l) << 16);
}

// ---- Pre-pass: weights -> WMMA B-fragment-ready f16 hi/lo. ----
// Entry t = ft*64 + kk*32 + lane. B frag, this code's k order (same as A): VGPR v holds K = 2v + 16*hf (+e),
// N(channel-in-tile) = lane%16.
__global__ __launch_bounds__(256) void prep_weights(const float* __restrict__ kw,
                                                    unsigned* __restrict__ wsB) {
    const int t = blockIdx.x * 256 + threadIdx.x;
    if (t >= 24 * 2 * 32) return;
    const int lane = t & 31;
    const int kk   = (t >> 5) & 1;
    const int ft   = t >> 6;
    const int hf   = lane >> 4;
    const int lr   = lane & 15;
    const float* krow = kw + (size_t)(ft * 16 + lr) * KSZ;

    v8u hv, lv;
#pragma unroll
    for (int v = 0; v < 8; ++v) {
        const int k0 = 32 * kk + 2 * v + 16 * hf;          // even pair start
        const float a0 = (k0     < KSZ) ? krow[k0]     : 0.0f;
        const float a1 = (k0 + 1 < KSZ) ? krow[k0 + 1] : 0.0f;
        const f16 h0 = (f16)a0, h1 = (f16)a1;
        const f16 l0 = (f16)(a0 - (float)h0), l1 = (f16)(a1 - (float)h1);
        hv[v] = (unsigned)__builtin_bit_cast(unsigned short, h0) |
                ((unsigned)__builtin_bit_cast(unsigned short, h1) << 16);
        lv[v] = (unsigned)__builtin_bit_cast(unsigned short, l0) |
                ((unsigned)__builtin_bit_cast(unsigned short, l1) << 16);
    }
    ((v8u*)wsB)[t]            = hv;
    ((v8u*)wsB)[t + WS_B_V8U] = lv;
}

// ---- Main kernel ----
__global__ __launch_bounds__(256) void spike_conv_wmma(
    const float* __restrict__ tj,     // [8,1,256,256]
    const unsigned* __restrict__ wsB, // pre-split weight fragments
    const float* __restrict__ Bv,     // [384,1]
    const float* __restrict__ Di,     // [9,384] (row 0)
    float* __restrict__ out)          // [8,384,256,256]
{
    __shared__ __align__(16) unsigned patchP[10 * 32]; // rows 0-6 real, 7-9 zero (K pad)
    __shared__ __align__(16) float    biasS[F_];

    const int tid = threadIdx.x;
    const int bid = blockIdx.x;
    const int xt = bid & 15;
    const int y  = (bid >> 4) & 255;
    const int b  = bid >> 12;
    const int x0 = xt << 4;

    // Stage packed {f16 hi|lo} patch; power-of-2 row pitch -> shift/mask only.
    for (int i = tid; i < 10 * 32; i += 256) {
        const int r = i >> 5, c = i & 31;
        const int yy = y + r - 3, xx = x0 + c - 3;
        float v = 0.0f;
        if (r < 7 && c < 22 && (unsigned)yy < (unsigned)H_ && (unsigned)xx < (unsigned)W_)
            v = tj[((size_t)b * H_ + yy) * W_ + xx];
        patchP[i] = pack_hilo(v);
    }
    // Fused bias: 1 - B[f] - D_i[0][f]
    for (int f = tid; f < F_; f += 256)
        biasS[f] = 1.0f - Bv[f] - Di[f];
    __syncthreads();

    const int lane = tid & 31;
    const int wave = tid >> 5;
    const int hf   = lane >> 4;
    const int lr   = lane & 15;

    // ---- A fragments (patch): ds_load_b32 (const offsets off lr) + v_perm. ----
    // A 16x32 frag, this code's k order (same as B): M(pixel)=lane%16; VGPR v -> K=2v+16*hf(+e).
    const unsigned* prow = patchP + lr;
    v8u ahw[2], alw[2];
#pragma unroll
    for (int kk = 0; kk < 2; ++kk) {
#pragma unroll
        for (int v = 0; v < 8; ++v) {
            const int k0 = 32 * kk + 2 * v + 16 * hf;
            const int k1 = k0 + 1;
            const int o0 = (k0 / KW_) * 32 + (k0 % KW_);  // compile-time consts
            const int o1 = (k1 / KW_) * 32 + (k1 % KW_);
            const unsigned d0 = prow[o0];
            const unsigned d1 = prow[o1];
            ahw[kk][v] = __builtin_amdgcn_perm(d1, d0, 0x05040100u); // hi halves
            alw[kk][v] = __builtin_amdgcn_perm(d1, d0, 0x07060302u); // lo halves
        }
    }

    // Per-wave bases (64-bit math once; per-tile deltas become store immediates).
    const v8u* bHi = ((const v8u*)wsB) + (size_t)wave * 192 + lane;   // + t*64 + kk*32
    const v8u* bLo = bHi + WS_B_V8U;
    const float* biasBase = biasS + wave * 48 + lr;                   // + t*16
    float* outBase = out + (((size_t)b * F_ + wave * 48 + lr) * H_ + y) * W_ + x0 + 8 * hf;

#pragma unroll
    for (int t = 0; t < 3; ++t) {
        // Bias pre-loaded into the accumulator (replaces the zero-init movs).
        const float bias = biasBase[t * 16];
        v8f acc = {bias, bias, bias, bias, bias, bias, bias, bias};

#pragma unroll
        for (int kk = 0; kk < 2; ++kk) {
            const v16h ahi = __builtin_bit_cast(v16h, ahw[kk]);
            const v16h alo = __builtin_bit_cast(v16h, alw[kk]);
            const v16h bhi = __builtin_bit_cast(v16h, bHi[t * 64 + kk * 32]);
            const v16h blo = __builtin_bit_cast(v16h, bLo[t * 64 + kk * 32]);
            acc = __builtin_amdgcn_wmma_f32_16x16x32_f16(false, ahi, false, bhi, (short)0, acc, false, false);
            acc = __builtin_amdgcn_wmma_f32_16x16x32_f16(false, ahi, false, blo, (short)0, acc, false, false);
            acc = __builtin_amdgcn_wmma_f32_16x16x32_f16(false, alo, false, bhi, (short)0, acc, false, false);
        }

        // D layout: VGPR r -> M(pixel) = r + 8*hf, N(channel) = lane%16.
        // Lane writes 8 consecutive floats of one channel row -> 2x b128 stores.
        float* orow = outBase + (size_t)t * 16 * (H_ * W_);
        const float4 s0 = {fminf(acc[0], 1.0f), fminf(acc[1], 1.0f),
                           fminf(acc[2], 1.0f), fminf(acc[3], 1.0f)};
        const float4 s1 = {fminf(acc[4], 1.0f), fminf(acc[5], 1.0f),
                           fminf(acc[6], 1.0f), fminf(acc[7], 1.0f)};
        *(float4*)(orow)     = s0;   // 32B-aligned (x0%16==0, 8*hf%8==0)
        *(float4*)(orow + 4) = s1;
    }
}

extern "C" void kernel_launch(void* const* d_in, const int* in_sizes, int n_in,
                              void* d_out, int out_size, void* d_ws, size_t ws_size,
                              hipStream_t stream) {
    (void)in_sizes; (void)n_in; (void)out_size; (void)ws_size;
    const float* tj   = (const float*)d_in[0];
    const float* kern = (const float*)d_in[1];
    const float* Bv   = (const float*)d_in[2];
    const float* Di   = (const float*)d_in[3];
    float* out = (float*)d_out;
    unsigned* wsB = (unsigned*)d_ws;   // needs 96 KB

    prep_weights<<<dim3(6), dim3(256), 0, stream>>>(kern, wsB);
    // One block per (batch, row, 16-pixel strip): 8 * 256 * 16 = 32768 blocks.
    spike_conv_wmma<<<dim3(8 * 256 * 16), dim3(256), 0, stream>>>(tj, wsB, Bv, Di, out);
}
